// TestModule_63239098466877
// MI455X (gfx1250) — matrix operand order repaired and checked by reading, not compiled, not run on hardware
//
#include <hip/hip_runtime.h>
#include <hip/hip_bf16.h>

// ---------------------------------------------------------------------------
// out[1024,95000] = concat(emb[1024,758], pad[1024,10]) @ W_rev[768,95000]
//
// MI455X plan: bf16 WMMA (f32 accum) GEMM, N-partitioned so W_rev streams from
// HBM exactly once (non-temporal); A (1.5MB bf16, WMMA-A-swizzled) lives in L2
// and is pumped into double-buffered LDS with CDNA5 async-to-LDS copies that
// overlap the WMMA chain; A-frag LDS reads ping-pong around the WMMAs.
// Output stores are non-temporal (write-once stream).
// ---------------------------------------------------------------------------

#define CTX        1024
#define EFFECT     758
#define ADD_DIM    10
#define EMBED      768      // EFFECT + ADD_DIM
#define KSYN       8
#define NVOCAB     95000
#define KT_COUNT   24       // 768 / 32
#define MB_COUNT   64       // 1024 / 16
#define WAVES_PER_WG 8
#define GEMM_GRID  743      // ceil(ceil(95000/16)/8)
#define TILE_BYTES (KT_COUNT * 1024)   // 24 KB per 16x768 A tile

typedef __bf16 bf16_t;
typedef __attribute__((ext_vector_type(16))) __bf16 v16bf;
typedef __attribute__((ext_vector_type(8)))  float  v8f;

__device__ __forceinline__ unsigned short f32_to_bf16_rne(float f) {
    unsigned u = __builtin_bit_cast(unsigned, f);
    u += 0x7FFFu + ((u >> 16) & 1u);   // round-to-nearest-even
    return (unsigned short)(u >> 16);
}

// Swizzled A layout per ISA 7.12.2 (16-bit A-matrix 16x32, wave32):
//   frag = (m/16)*24 + (k/32); lane = ((k%32/8)&1)*16 + (m%16)
//   u16  = (k%32 >= 16 ? 8 : 0) + (k%32 & 7)
__device__ __forceinline__ size_t a_swz_index(int m, int k) {
    int k2   = k & 31;
    int h    = (k2 >> 3) & 1;
    int off  = ((k2 >= 16) ? 8 : 0) + (k2 & 7);
    int lane = h * 16 + (m & 15);
    size_t frag = (size_t)((m >> 4) * KT_COUNT + (k >> 5));
    return frag * 512 + (size_t)lane * 16 + off;
}

// ---------------------------------------------------------------------------
// Phase 1: one block per token. base + masked synonym accumulation, append
// padding, write bf16 directly into WMMA-A swizzled workspace.
// ---------------------------------------------------------------------------
__global__ void __launch_bounds__(256) emb_kernel(
        const int*   __restrict__ ids,
        const float* __restrict__ W_emb,
        const float* __restrict__ padding,
        const int*   __restrict__ syn_table,
        const int*   __restrict__ syn_mask,
        unsigned short* __restrict__ A_swz) {
    const int m   = blockIdx.x;
    const int tid = threadIdx.x;
    const int id  = ids[m];

    const float* base = W_emb + (size_t)id * EFFECT;
    const float* syn[KSYN];
    float msk[KSYN];
#pragma unroll
    for (int k = 0; k < KSYN; ++k) {
        int sid = syn_table[id * KSYN + k];
        syn[k]  = W_emb + (size_t)sid * EFFECT;
        msk[k]  = (float)syn_mask[id * KSYN + k];
    }

    float ps[KSYN + 1];
#pragma unroll
    for (int s = 0; s <= KSYN; ++s) ps[s] = 0.0f;
    for (int j = tid; j < EFFECT; j += 256) {
        ps[0] += base[j];
#pragma unroll
        for (int k = 0; k < KSYN; ++k) ps[k + 1] += syn[k][j];
    }

    __shared__ float red[256];
    __shared__ float sums[KSYN + 1];
#pragma unroll 1
    for (int s = 0; s <= KSYN; ++s) {
        red[tid] = ps[s];
        __syncthreads();
        for (int off = 128; off > 0; off >>= 1) {
            if (tid < off) red[tid] += red[tid + off];
            __syncthreads();
        }
        if (tid == 0) sums[s] = red[0];
        __syncthreads();
    }

    const float i_sum = sums[0];
    float coef[KSYN];
#pragma unroll
    for (int k = 0; k < KSYN; ++k) coef[k] = msk[k] * (i_sum / sums[k + 1]);

    for (int j = tid; j < EMBED; j += 256) {
        float v;
        if (j < EFFECT) {
            v = base[j];
#pragma unroll
            for (int k = 0; k < KSYN; ++k) v += coef[k] * syn[k][j];
        } else {
            v = padding[m * ADD_DIM + (j - EFFECT)];
        }
        A_swz[a_swz_index(m, j)] = f32_to_bf16_rne(v);
    }
}

// ---------------------------------------------------------------------------
// CDNA5 async copy: global -> LDS, b128, no VGPR staging, ASYNCcnt-tracked.
// ---------------------------------------------------------------------------
__device__ __forceinline__ void async_copy_b128(unsigned lds_off,
                                                const void* gaddr) {
    asm volatile("global_load_async_to_lds_b128 %0, %1, off"
                 :
                 : "v"(lds_off), "v"(gaddr)
                 : "memory");
}
__device__ __forceinline__ void wait_asynccnt0() {
    asm volatile("s_wait_asynccnt 0x0" ::: "memory");
}

// ---------------------------------------------------------------------------
// Phase 2 GEMM: WG = 8 waves = 128 columns, all 1024 rows.
// B panel (768x16 per wave) resident as 24 v16bf frags; A tiles async-copied
// into double-buffered LDS; A-frag LDS reads ping-pong (2 live frags) so each
// WMMA has a full WMMA of LDS-load slack.
// ---------------------------------------------------------------------------
__global__ void __launch_bounds__(256, 1) gemm_kernel(
        const float* __restrict__ Wrev,
        const unsigned short* __restrict__ A_swz,
        float* __restrict__ out) {
    const int tid  = threadIdx.x;
    const int wave = tid >> 5;
    const int lane = tid & 31;
    const int h    = lane >> 4;   // K-half selector (B & C layouts)
    const int nl   = lane & 15;   // column within 16-wide tile

    const int n    = (blockIdx.x * WAVES_PER_WG + wave) * 16 + nl;
    const bool nok = (n < NVOCAB);

    // --- Resident B fragments (ISA 7.12.2 B 32x16: lane = 16h + col,
    //     packed bf16 t -> K = 32*kt + 16*(t/8) + 8h + t%8). W_rev streamed NT, once. ---
    v16bf bfrag[KT_COUNT];
#pragma unroll
    for (int kt = 0; kt < KT_COUNT; ++kt) {
        union { unsigned short u[16]; v16bf v; } tmp;
#pragma unroll
        for (int t = 0; t < 16; ++t) {
            int kk  = kt * 32 + (t >> 3) * 16 + h * 8 + (t & 7);
            float f = nok ? __builtin_nontemporal_load(&Wrev[(size_t)kk * NVOCAB + n])
                          : 0.0f;
            tmp.u[t] = f32_to_bf16_rne(f);
        }
        bfrag[kt] = tmp.v;
    }

    // Double-buffered A tile in LDS (2 x 24 KB).
    __shared__ __align__(32) unsigned char alds[2 * TILE_BYTES];
    const unsigned lds_base = (unsigned)(unsigned long long)(void*)&alds[0];

    // Per-thread async copy of one 24KB tile: 6 x b128, fully coalesced.
    const unsigned char* a_bytes = (const unsigned char*)A_swz;
    auto issue_tile = [&](int mb, int buf) {
        const unsigned char* src = a_bytes + (size_t)mb * TILE_BYTES + tid * 16;
        unsigned dst = lds_base + (unsigned)buf * TILE_BYTES + tid * 16;
#pragma unroll
        for (int i = 0; i < 6; ++i)
            async_copy_b128(dst + i * 4096, src + i * 4096);
    };

    // Prologue: fill buffer 0 with tile 0.
    issue_tile(0, 0);
    wait_asynccnt0();
    __syncthreads();

#pragma unroll 1
    for (int mb = 0; mb < MB_COUNT; ++mb) {
        const int cur = mb & 1;
        if (mb + 1 < MB_COUNT) issue_tile(mb + 1, cur ^ 1);   // overlap with WMMA

        const unsigned char* abuf = &alds[cur * TILE_BYTES];
        auto lda = [&](int kt) -> v16bf {
            return *(const v16bf*)(abuf + kt * 1024 + lane * 32);
        };

        v8f c = {};
        // Explicit 2-frag ping-pong: wmma(aX, kt) precedes the reload of aX
        // for kt+2, forcing two live A-frag register sets (no WAR collapse).
        v16bf a0 = lda(0);
        v16bf a1 = lda(1);
#pragma unroll
        for (int kt = 0; kt < KT_COUNT; kt += 2) {
            c = __builtin_amdgcn_wmma_f32_16x16x32_bf16(
                    false, a0, false, bfrag[kt], (short)0, c, false, false);
            if (kt + 2 < KT_COUNT) a0 = lda(kt + 2);
            c = __builtin_amdgcn_wmma_f32_16x16x32_bf16(
                    false, a1, false, bfrag[kt + 1], (short)0, c, false, false);
            if (kt + 3 < KT_COUNT) a1 = lda(kt + 3);
        }

        // C/D layout: lane = 16h + col; VGPR r holds row (8h + r). NT stores.
#pragma unroll
        for (int r = 0; r < 8; ++r) {
            int mm = mb * 16 + h * 8 + r;
            if (nok) __builtin_nontemporal_store(c[r], &out[(size_t)mm * NVOCAB + n]);
        }

        wait_asynccnt0();   // own copies of next tile landed
        __syncthreads();    // everyone's copies landed; reads of cur done
    }
}

// ---------------------------------------------------------------------------
extern "C" void kernel_launch(void* const* d_in, const int* in_sizes, int n_in,
                              void* d_out, int out_size, void* d_ws, size_t ws_size,
                              hipStream_t stream) {
    const int*   ids       = (const int*)  d_in[0];
    const float* W_emb     = (const float*)d_in[1];
    const float* W_rev     = (const float*)d_in[2];
    const float* padding   = (const float*)d_in[3];
    const int*   syn_table = (const int*)  d_in[4];
    const int*   syn_mask  = (const int*)  d_in[5];

    unsigned short* A_swz = (unsigned short*)d_ws;  // 1024*768*2 = 1.5 MB

    emb_kernel<<<CTX, 256, 0, stream>>>(ids, W_emb, padding, syn_table, syn_mask, A_swz);
    gemm_kernel<<<GEMM_GRID, 256, 0, stream>>>(W_rev, A_swz, (float*)d_out);
}
